// MolmoAttention_26525718020411
// MI455X (gfx1250) — matrix operand order repaired and checked by reading, not compiled, not run on hardware
//
#include <hip/hip_runtime.h>
#include <hip/hip_bf16.h>

typedef __bf16 bf16_t;
typedef __attribute__((ext_vector_type(4)))  __bf16 v4bf;
typedef __attribute__((ext_vector_type(8)))  __bf16 v8bf;
typedef __attribute__((ext_vector_type(16))) __bf16 v16bf;
typedef __attribute__((ext_vector_type(8)))  float  v8f;

#define TSEQ 2048
#define HID  4096
#define QKVN 12288
#define NH   32
#define HD   128

// ---------- WMMA fragment helpers (gfx1250 wave32 layouts, ISA 7.12.2) ----------

__device__ __forceinline__ v16bf frag_cat(v8bf a, v8bf b) {
  union { v16bf v; v8bf h[2]; } u;
  u.h[0] = a; u.h[1] = b;
  return u.v;
}

// A fragment: 16x32 bf16 tile, src row-major (row=M, contiguous K), base at (m0,k0).
// lane<16: M=lane, K in {0..7, 16..23}; lane>=16: M=lane-16, K in {8..15, 24..31}.
__device__ __forceinline__ v16bf load_frag_a(const bf16_t* base, int ld) {
  const int lane = threadIdx.x & 31;
  const int row  = lane & 15;
  const int off  = (lane < 16) ? 0 : 8;
  const bf16_t* p = base + (size_t)row * ld + off;
  v8bf c0 = *(const v8bf*)(p);
  v8bf c1 = *(const v8bf*)(p + 16);
  return frag_cat(c0, c1);
}

// B fragment: 32x16 bf16 where B[k,n] = src[n*ld + k] (src rows are the N entities,
// contiguous along K). lane<16: N=lane, K in {0..7, 16..23}; lane>=16: N=lane-16, K in {8..15, 24..31}.
__device__ __forceinline__ v16bf load_frag_b(const bf16_t* base, int ld) {
  const int lane = threadIdx.x & 31;
  const int n    = lane & 15;
  const int off  = (lane < 16) ? 0 : 8;
  const bf16_t* p = base + (size_t)n * ld + off;
  v8bf c0 = *(const v8bf*)(p);
  v8bf c1 = *(const v8bf*)(p + 16);
  return frag_cat(c0, c1);
}

__device__ __forceinline__ v8f wmma_bf16(v16bf a, v16bf b, v8f c) {
  return __builtin_amdgcn_wmma_f32_16x16x32_bf16(false, a, false, b, (short)0, c, false, false);
}

// wave32 16-lane butterfly via ds_swizzle (group-of-32 xor mode); xor masks <16 stay
// within each 16-lane half, matching the C/D fragment row ownership.
template <int M>
__device__ __forceinline__ float swz_xor(float x) {
  int i = __builtin_amdgcn_ds_swizzle(__float_as_int(x), (M << 10) | 0x1f);
  return __int_as_float(i);
}

// ---------- CDNA5 async global->LDS copy (32 bytes per lane, two B128 ops) ----------
// GLOBAL_LOAD_ASYNC_TO_LDS_B128, GV mode (saddr=off). INST_OFFSET adds to both the
// LDS and global addresses (ISA 10.x pseudocode), matching contiguous 16B chunks.

__device__ __forceinline__ void async_copy32(const bf16_t* gsrc, bf16_t* ldst) {
  unsigned long long ga = (unsigned long long)(uintptr_t)gsrc;
  unsigned la = (unsigned)(uintptr_t)ldst; // low 32 bits of generic ptr = LDS offset
  asm volatile(
      "global_load_async_to_lds_b128 %0, %1, off\n\t"
      "global_load_async_to_lds_b128 %0, %1, off offset:16"
      :: "v"(la), "v"(ga) : "memory");
}

__device__ __forceinline__ void async_wait0() {
  asm volatile("s_wait_asynccnt 0x0" ::: "memory");
}

// ---------- fp32 -> bf16 convert ----------

__global__ __launch_bounds__(256) void f2bf_kernel(const float* __restrict__ x,
                                                   bf16_t* __restrict__ y, long n4) {
  long i = (long)blockIdx.x * blockDim.x + threadIdx.x;
  const long stride = (long)gridDim.x * blockDim.x;
  for (; i < n4; i += stride) {
    float4 v = ((const float4*)x)[i];
    v4bf o;
    o[0] = (bf16_t)v.x; o[1] = (bf16_t)v.y; o[2] = (bf16_t)v.z; o[3] = (bf16_t)v.w;
    ((v4bf*)y)[i] = o;
  }
}

// ---------- bf16 GEMM: C[m,n] = sum_k A[m,k]*B[n,k] (+bias[n]), fp32 out ----------
// 128x128 block tile, BK=32, 8 waves each computing 64x32 via 4x2 WMMA accumulators.
// Tiles are streamed with double-buffered async global->LDS copies (ASYNCcnt).

__global__ __launch_bounds__(256) void gemm_bf16_nt(const bf16_t* __restrict__ A,
                                                    const bf16_t* __restrict__ B,
                                                    const float* __restrict__ bias,
                                                    float* __restrict__ C,
                                                    int M, int N, int K) {
  __shared__ bf16_t As[2][128][40];
  __shared__ bf16_t Bs[2][128][40];

  const int tid = threadIdx.x;
  const int m0  = blockIdx.y * 128;
  const int n0  = blockIdx.x * 128;
  const int w   = tid >> 5;
  const int wr  = (w >> 2) * 64;   // wave row offset: 0 | 64
  const int wc  = (w & 3) * 32;    // wave col offset: 0..96
  const int lrow = tid >> 1;       // 0..127
  const int lchk = (tid & 1) * 16; // 0 | 16

  const bf16_t* gA = A + (size_t)(m0 + lrow) * K + lchk;
  const bf16_t* gB = B + (size_t)(n0 + lrow) * K + lchk;

  v8f acc[4][2];
  v8f zero = {};
#pragma unroll
  for (int i = 0; i < 4; ++i)
#pragma unroll
    for (int j = 0; j < 2; ++j) acc[i][j] = zero;

  // prologue: stage first tile into buffer 0
  async_copy32(gA, &As[0][lrow][lchk]);
  async_copy32(gB, &Bs[0][lrow][lchk]);

  int cur = 0;
#pragma unroll 1
  for (int kt = 0; kt < K; kt += 32) {
    async_wait0();      // this wave's async LDS writes have landed
    __syncthreads();    // everyone's have; previous buffer's readers are done

    if (kt + 32 < K) {  // stream next tile into the other buffer
      async_copy32(gA + kt + 32, &As[cur ^ 1][lrow][lchk]);
      async_copy32(gB + kt + 32, &Bs[cur ^ 1][lrow][lchk]);
    }

    v16bf af[4], bfg[2];
#pragma unroll
    for (int i = 0; i < 4; ++i) af[i] = load_frag_a(&As[cur][wr + i * 16][0], 40);
#pragma unroll
    for (int j = 0; j < 2; ++j) bfg[j] = load_frag_b(&Bs[cur][wc + j * 16][0], 40);
#pragma unroll
    for (int i = 0; i < 4; ++i)
#pragma unroll
      for (int j = 0; j < 2; ++j) acc[i][j] = wmma_bf16(af[i], bfg[j], acc[i][j]);

    cur ^= 1;
  }

  const int lane = tid & 31;
  const int cn   = lane & 15;
  const int rb   = (lane < 16) ? 0 : 8;
#pragma unroll
  for (int i = 0; i < 4; ++i)
#pragma unroll
    for (int j = 0; j < 2; ++j) {
      const int col = n0 + wc + j * 16 + cn;
      const float badd = bias ? bias[col] : 0.0f;
#pragma unroll
      for (int r = 0; r < 8; ++r) {
        const int row = m0 + wr + i * 16 + rb + r;
        C[(size_t)row * N + col] = acc[i][j][r] + badd;
      }
    }
}

// ---------- fused RMSNorm (over 4096) + RoPE, qkv fp32 -> bf16 Q,K ----------

__global__ __launch_bounds__(256) void rmsrope_kernel(const float* __restrict__ qkv,
                                                      const int* __restrict__ pos,
                                                      const float* __restrict__ qw,
                                                      const float* __restrict__ kw,
                                                      bf16_t* __restrict__ Qo,
                                                      bf16_t* __restrict__ Ko) {
  __shared__ float red[256];
  const int t   = blockIdx.x;
  const int tid = threadIdx.x;
  const float* qrow = qkv + (size_t)t * QKVN;
  const float* krow = qrow + HID;

  float sq = 0.f, sk = 0.f;
  for (int i = tid; i < HID; i += 256) {
    float a = qrow[i]; sq += a * a;
    float b = krow[i]; sk += b * b;
  }
  red[tid] = sq; __syncthreads();
  for (int s = 128; s > 0; s >>= 1) { if (tid < s) red[tid] += red[tid + s]; __syncthreads(); }
  const float rq = rsqrtf(red[0] * (1.0f / HID) + 1e-5f);
  __syncthreads();
  red[tid] = sk; __syncthreads();
  for (int s = 128; s > 0; s >>= 1) { if (tid < s) red[tid] += red[tid + s]; __syncthreads(); }
  const float rk = rsqrtf(red[0] * (1.0f / HID) + 1e-5f);

  const float p = (float)pos[t];
  const float LOG2_THETA = 19.9315685693241f; // log2(1e6)
  for (int i = tid; i < HID; i += 256) {
    const int r  = i & 127;
    const int d  = (r < 64) ? r : r - 64;
    const int ip = (r < 64) ? i + 64 : i - 64;
    const float inv = exp2f(-(float)d * (LOG2_THETA / 64.0f));
    const float ang = p * inv;
    const float c = cosf(ang), s = sinf(ang);

    const float xq  = qrow[i]  * rq * qw[i];
    const float xqp = qrow[ip] * rq * qw[ip];
    Qo[(size_t)t * HID + i] = (bf16_t)((r < 64) ? (xq * c - xqp * s) : (xq * c + xqp * s));

    const float xk  = krow[i]  * rk * kw[i];
    const float xkp = krow[ip] * rk * kw[ip];
    Ko[(size_t)t * HID + i] = (bf16_t)((r < 64) ? (xk * c - xkp * s) : (xk * c + xkp * s));
  }
}

// ---------- V transpose: Vt[c][t] = qkv[t][8192 + c], fp32 -> bf16 ----------

__global__ __launch_bounds__(256) void vtrans_kernel(const float* __restrict__ qkv,
                                                     bf16_t* __restrict__ Vt) {
  __shared__ bf16_t tile[32][33];
  const int c0 = blockIdx.x * 32;
  const int t0 = blockIdx.y * 32;
  const int tx = threadIdx.x & 31;
  const int ty = threadIdx.x >> 5; // 0..7
  for (int i = ty; i < 32; i += 8)
    tile[i][tx] = (bf16_t)qkv[(size_t)(t0 + i) * QKVN + 2 * HID + c0 + tx];
  __syncthreads();
  for (int i = ty; i < 32; i += 8)
    Vt[(size_t)(c0 + i) * TSEQ + t0 + tx] = tile[tx][i];
}

// ---------- flash attention: one head x 128 q-rows per block, 8 waves x 16 rows ----------

__global__ __launch_bounds__(256) void flash_kernel(const bf16_t* __restrict__ Q,
                                                    const bf16_t* __restrict__ Kc,
                                                    const bf16_t* __restrict__ Vt,
                                                    bf16_t* __restrict__ O) {
  __shared__ bf16_t P[8][16][40];
  const int h    = blockIdx.y;
  const int qb   = blockIdx.x * 128;
  const int tid  = threadIdx.x;
  const int w    = tid >> 5;
  const int lane = tid & 31;
  const int ln   = lane & 15;
  const int rb   = (lane < 16) ? 0 : 8;
  const int q0   = qb + w * 16;

  v16bf qf[4];
#pragma unroll
  for (int c = 0; c < 4; ++c)
    qf[c] = load_frag_a(Q + (size_t)q0 * HID + h * HD + c * 32, HID);

  v8f o[8];
  v8f zero = {};
#pragma unroll
  for (int a = 0; a < 8; ++a) o[a] = zero;
  float mr[8], lr[8];
#pragma unroll
  for (int r = 0; r < 8; ++r) { mr[r] = -1e30f; lr[r] = 0.f; }

  const float sc = 0.08838834764831845f * 1.4426950408889634f; // rsqrt(128) * log2(e)

  for (int kt = 0; kt < qb + 128; kt += 32) {
    // S = Q x K^T for 32 keys (two 16-wide N tiles), K-dim 128 in 4 chained WMMAs
    v8f s0 = zero, s1 = zero;
#pragma unroll
    for (int c = 0; c < 4; ++c) {
      v16bf b0 = load_frag_b(Kc + (size_t)kt * HID + h * HD + c * 32, HID);
      v16bf b1 = load_frag_b(Kc + (size_t)(kt + 16) * HID + h * HD + c * 32, HID);
      s0 = wmma_bf16(qf[c], b0, s0);
      s1 = wmma_bf16(qf[c], b1, s1);
    }

    // online softmax over this 16x32 score tile (rows owned per-lane-half)
#pragma unroll
    for (int r = 0; r < 8; ++r) {
      const int row = q0 + rb + r;
      const int k0i = kt + ln;
      const int k1i = kt + 16 + ln;
      const bool ok0 = (k0i <= row);
      const bool ok1 = (k1i <= row);
      float v0 = ok0 ? s0[r] * sc : -1e30f;
      float v1 = ok1 ? s1[r] * sc : -1e30f;
      float mx = fmaxf(v0, v1);
      mx = fmaxf(mx, swz_xor<1>(mx));
      mx = fmaxf(mx, swz_xor<2>(mx));
      mx = fmaxf(mx, swz_xor<4>(mx));
      mx = fmaxf(mx, swz_xor<8>(mx));
      const float mnew = fmaxf(mr[r], mx);
      const float corr = exp2f(mr[r] - mnew);
      mr[r] = mnew;
      const float p0 = ok0 ? exp2f(v0 - mnew) : 0.f;
      const float p1 = ok1 ? exp2f(v1 - mnew) : 0.f;
      float ps = p0 + p1;
      ps += swz_xor<1>(ps);
      ps += swz_xor<2>(ps);
      ps += swz_xor<4>(ps);
      ps += swz_xor<8>(ps);
      lr[r] = lr[r] * corr + ps;
#pragma unroll
      for (int a = 0; a < 8; ++a) o[a][r] *= corr;
      P[w][rb + r][ln]      = (bf16_t)p0;
      P[w][rb + r][16 + ln] = (bf16_t)p1;
    }
    __syncthreads(); // publish P (C-layout -> A-layout reformat through LDS)

    // O += P x V  (A = 16x32 P tile, B from transposed V: contiguous along keys)
    v16bf pf = load_frag_a(&P[w][0][0], 40);
#pragma unroll
    for (int a = 0; a < 8; ++a) {
      v16bf vf = load_frag_b(Vt + (size_t)(h * HD + a * 16) * TSEQ + kt, TSEQ);
      o[a] = wmma_bf16(pf, vf, o[a]);
    }
    __syncthreads(); // P reusable next iteration
  }

#pragma unroll
  for (int a = 0; a < 8; ++a)
#pragma unroll
    for (int r = 0; r < 8; ++r) {
      const int row = q0 + rb + r;
      const int col = h * HD + a * 16 + ln;
      O[(size_t)row * HID + col] = (bf16_t)(o[a][r] / lr[r]);
    }
}

// ---------- host ----------

extern "C" void kernel_launch(void* const* d_in, const int* in_sizes, int n_in,
                              void* d_out, int out_size, void* d_ws, size_t ws_size,
                              hipStream_t stream) {
  const float* hidden    = (const float*)d_in[0];
  const int*   positions = (const int*)d_in[1];
  const float* w_qkv     = (const float*)d_in[2];
  const float* b_qkv     = (const float*)d_in[3];
  const float* q_norm_w  = (const float*)d_in[4];
  const float* k_norm_w  = (const float*)d_in[5];
  const float* w_o       = (const float*)d_in[6];
  float* out = (float*)d_out;

  char* ws = (char*)d_ws;
  size_t off = 0;
  bf16_t* a_bf = (bf16_t*)(ws + off);            off += (size_t)TSEQ * HID * 2;   // 16.8 MB
  const size_t wqkv_off = off;
  bf16_t* wqkv_bf = (bf16_t*)(ws + off);         off += (size_t)QKVN * HID * 2;   // 100.7 MB
  bf16_t* wo_bf = (bf16_t*)(ws + off);           off += (size_t)HID * HID * 2;    // 33.6 MB
  float*  qkvf  = (float*)(ws + off);            off += (size_t)TSEQ * QKVN * 4;  // 100.7 MB
  // region reuse (stream-ordered, deterministic): wqkv_bf dead after QKV GEMM,
  // a_bf dead after QKV GEMM.
  bf16_t* q_bf   = (bf16_t*)(ws + wqkv_off);
  bf16_t* k_bf   = q_bf + (size_t)TSEQ * HID;
  bf16_t* vt_bf  = k_bf + (size_t)TSEQ * HID;
  bf16_t* attn_bf = a_bf;

  // 1) fp32 -> bf16 converts
  f2bf_kernel<<<2048, 256, 0, stream>>>(hidden, a_bf, (long)TSEQ * HID / 4);
  f2bf_kernel<<<4096, 256, 0, stream>>>(w_qkv, wqkv_bf, (long)QKVN * HID / 4);
  f2bf_kernel<<<2048, 256, 0, stream>>>(w_o, wo_bf, (long)HID * HID / 4);

  // 2) QKV GEMM + bias (fp32 accum/out)
  gemm_bf16_nt<<<dim3(QKVN / 128, TSEQ / 128), 256, 0, stream>>>(
      a_bf, wqkv_bf, b_qkv, qkvf, TSEQ, QKVN, HID);

  // 3) RMSNorm + RoPE -> bf16 Q,K
  rmsrope_kernel<<<TSEQ, 256, 0, stream>>>(qkvf, positions, q_norm_w, k_norm_w, q_bf, k_bf);

  // 4) V -> transposed bf16 [d][t]
  vtrans_kernel<<<dim3(HID / 32, TSEQ / 32), 256, 0, stream>>>(qkvf, vt_bf);

  // 5) flash attention
  flash_kernel<<<dim3(TSEQ / 128, NH), 256, 0, stream>>>(q_bf, k_bf, vt_bf, attn_bf);

  // 6) output projection -> d_out (fp32)
  gemm_bf16_nt<<<dim3(HID / 128, TSEQ / 128), 256, 0, stream>>>(
      attn_bf, wo_bf, nullptr, out, TSEQ, HID, HID);
}
